// ViewLearner_58128087384892
// MI455X (gfx1250) — matrix operand order repaired and checked by reading, not compiled, not run on hardware
//
#include <hip/hip_runtime.h>
#include <hip/hip_bf16.h>
#include <math.h>

typedef __attribute__((ext_vector_type(16))) _Float16 v16h;
typedef __attribute__((ext_vector_type(8)))  float    v8f;

#define D_IN   128
#define KSTEP  32
#define NK     8                     // 256 / 32 K-steps
#define NN     8                     // 128 / 16 N-tiles
#define SLOTS  (NK * NN * 32)        // 2048 lane-slots of 16 halfs
#define W1HALFS (SLOTS * 16)         // 32768 halfs = 64 KB packed W1

// ---------------------------------------------------------------------------
// Pack W1 (256x128 f32 row-major) into wave32 WMMA B-fragment order as f16.
// Fragment (kk,n): B tile = W1[kk*32 .. kk*32+31][n*16 .. n*16+15].
// Lane L holds column n*16 + (L&15); lanes 0-15 cover K=kk*32+0..7,16..23,
// lanes 16-31 cover K=kk*32+8..15,24..31 (v16h).
// ---------------------------------------------------------------------------
__global__ void pack_w1_kernel(const float* __restrict__ W1,
                               _Float16* __restrict__ w1p)
{
    int s    = blockIdx.x * blockDim.x + threadIdx.x;   // slot 0..2047
    int lane = s & 31;
    int frag = s >> 5;                                  // kk*8 + n
    int kk   = frag >> 3;
    int n    = frag & 7;
    int kbase = kk * KSTEP + ((lane >> 4) << 3);
    int col   = n * 16 + (lane & 15);
    union { v16h v; _Float16 h[16]; } u;
#pragma unroll
    for (int kl = 0; kl < 16; ++kl)
        u.h[kl] = (_Float16)W1[(size_t)(kbase + kl + ((kl >> 3) << 3)) * 128 + col];
    *(v16h*)(w1p + (size_t)s * 16) = u.v;
}

// Build one A fragment (v16h) from 32 consecutive f32 at rp + hi*8:
// lane half lo: K 0..7 & 16..23; lane half hi: K 8..15 & 24..31.
__device__ __forceinline__ v16h load_a_frag(const float* p0)
{
    float4 f0 = *(const float4*)(p0);
    float4 f1 = *(const float4*)(p0 + 4);
    float4 g0 = *(const float4*)(p0 + 16);
    float4 g1 = *(const float4*)(p0 + 20);
    union { v16h v; _Float16 h[16]; } a;
    a.h[0]  = (_Float16)f0.x; a.h[1]  = (_Float16)f0.y;
    a.h[2]  = (_Float16)f0.z; a.h[3]  = (_Float16)f0.w;
    a.h[4]  = (_Float16)f1.x; a.h[5]  = (_Float16)f1.y;
    a.h[6]  = (_Float16)f1.z; a.h[7]  = (_Float16)f1.w;
    a.h[8]  = (_Float16)g0.x; a.h[9]  = (_Float16)g0.y;
    a.h[10] = (_Float16)g0.z; a.h[11] = (_Float16)g0.w;
    a.h[12] = (_Float16)g1.x; a.h[13] = (_Float16)g1.y;
    a.h[14] = (_Float16)g1.z; a.h[15] = (_Float16)g1.w;
    return a.v;
}

// ReLU(C+b1) . W2 row partials + butterfly reduce over each 16-lane half.
__device__ __forceinline__ void reduce_tile(const v8f (&acc)[NN],
                                            const float (&b1v)[NN],
                                            const float (&w2v)[NN],
                                            float (&p)[8])
{
#pragma unroll
    for (int j = 0; j < 8; ++j) {
        float s = 0.f;
#pragma unroll
        for (int n = 0; n < NN; ++n)
            s += fmaxf(acc[n][j] + b1v[n], 0.f) * w2v[n];
        p[j] = s;
    }
#pragma unroll
    for (int m = 1; m < 16; m <<= 1) {
#pragma unroll
        for (int j = 0; j < 8; ++j)
            p[j] += __shfl_xor(p[j], m, 32);
    }
}

// Gumbel-sigmoid gate + RelaxedBernoulli sample for one edge.
__device__ __forceinline__ void emit_edge(long e, float logit,
                                          const float* __restrict__ u_gate,
                                          const float* __restrict__ u_bern,
                                          float* __restrict__ out, int n_edges)
{
    if (e >= n_edges) return;
    const float BIAS = 1e-4f;
    float ug   = u_gate[e];
    float eps  = (BIAS - (1.f - BIAS)) * ug + (1.f - BIAS);
    float gate = logf(eps) - log1pf(-eps) + logit;        // TEMP_GATE = 1
    float ew   = 1.f / (1.f + __expf(-gate));
    float att  = fminf(fmaxf(ew, 0.01f), 0.99f);
    float blog = logf(att) - log1pf(-att);
    float ub   = fminf(fmaxf(u_bern[e], 1e-7f), 1.f - 1e-7f);
    float z    = (blog + logf(ub) - log1pf(-ub)) * (1.f / 0.9f);
    float w    = 1.f / (1.f + __expf(-z));
    out[e] = (w > 0.f) ? w : 0.f;                         // mask
}

// ---------------------------------------------------------------------------
// Main fused kernel: gather -> f16 WMMA GEMM (256->128) -> ReLU -> dot(128->1)
// -> gumbel-sigmoid gate -> relaxed-bernoulli sample.
// Two 16-edge M-tiles per wave: every LDS B-fragment feeds two WMMAs.
// ---------------------------------------------------------------------------
__global__ __launch_bounds__(256) void edge_gate_kernel(
    const float*    __restrict__ node_emb,
    const int*      __restrict__ src,
    const int*      __restrict__ dst,
    const _Float16* __restrict__ w1p,
    const float*    __restrict__ b1,
    const float*    __restrict__ W2,
    const float*    __restrict__ b2,
    const float*    __restrict__ u_gate,
    const float*    __restrict__ u_bern,
    float*          __restrict__ out,
    int n_edges)
{
    __shared__ __align__(32) _Float16 ldsW1[W1HALFS];

    // Stage packed W1 (64 KB) into LDS with CDNA5 async global->LDS DMA:
    // no VGPR round-trip, tracked on ASYNCcnt.
    {
        for (int i = threadIdx.x; i < W1HALFS / 8; i += 256) {
            unsigned    lds_off = (unsigned)(size_t)(ldsW1 + (size_t)i * 8);
            const void* gptr    = (const void*)(w1p + (size_t)i * 8);
            asm volatile("global_load_async_to_lds_b128 %0, %1, off"
                         :: "v"(lds_off), "v"(gptr) : "memory");
        }
        asm volatile("s_wait_asynccnt 0" ::: "memory");
    }
    __syncthreads();

    const int  lane = threadIdx.x & 31;
    const int  wid  = threadIdx.x >> 5;
    const long e0   = ((long)blockIdx.x * 8 + wid) * 32;   // 32 edges per wave
    const int  r    = lane & 15;       // A row (edge within tile)
    const int  hi   = lane >> 4;       // lane half selects K sub-range
    const int  c    = lane & 15;       // C/D column for this lane

    long er0 = e0 + r;      if (er0 >= n_edges) er0 = n_edges - 1;
    long er1 = e0 + 16 + r; if (er1 >= n_edges) er1 = n_edges - 1;
    const float* s0 = node_emb + (size_t)src[er0] * D_IN;
    const float* d0 = node_emb + (size_t)dst[er0] * D_IN;
    const float* s1 = node_emb + (size_t)src[er1] * D_IN;
    const float* d1 = node_emb + (size_t)dst[er1] * D_IN;

    v8f acc0[NN], acc1[NN];
#pragma unroll
    for (int n = 0; n < NN; ++n) {
        acc0[n] = v8f{0.f, 0.f, 0.f, 0.f, 0.f, 0.f, 0.f, 0.f};
        acc1[n] = v8f{0.f, 0.f, 0.f, 0.f, 0.f, 0.f, 0.f, 0.f};
    }

#pragma unroll
    for (int kk = 0; kk < NK; ++kk) {
        // concat column window: kk<4 -> src half, kk>=4 -> dst half
        const float* rp0 = (kk < 4) ? (s0 + kk * KSTEP) : (d0 + (kk - 4) * KSTEP);
        const float* rp1 = (kk < 4) ? (s1 + kk * KSTEP) : (d1 + (kk - 4) * KSTEP);
        v16h a0 = load_a_frag(rp0 + hi * 8);
        v16h a1 = load_a_frag(rp1 + hi * 8);

#pragma unroll
        for (int n = 0; n < NN; ++n) {
            v16h bf = *(const v16h*)(ldsW1 + ((size_t)(kk * NN + n) * 32 + lane) * 16);
            acc0[n] = __builtin_amdgcn_wmma_f32_16x16x32_f16(
                false, a0, false, bf, (short)0, acc0[n], false, false);
            acc1[n] = __builtin_amdgcn_wmma_f32_16x16x32_f16(
                false, a1, false, bf, (short)0, acc1[n], false, false);
        }
    }

    // Second layer: logit[m] = sum_col relu(C[m,col] + b1[col]) * W2[col] + b2.
    float b1v[NN], w2v[NN];
#pragma unroll
    for (int n = 0; n < NN; ++n) {
        b1v[n] = b1[n * 16 + c];
        w2v[n] = W2[n * 16 + c];
    }
    const float b2v = b2[0];

    float p0[8], p1[8];
    reduce_tile(acc0, b1v, w2v, p0);
    reduce_tile(acc1, b1v, w2v, p1);

    // Lanes with c<8 finish edges: rows 0-7 (lo half), 8-15 (hi half), per tile.
    if (c < 8) {
        float l0 =
            (c & 4) ? ((c & 2) ? ((c & 1) ? p0[7] : p0[6]) : ((c & 1) ? p0[5] : p0[4]))
                    : ((c & 2) ? ((c & 1) ? p0[3] : p0[2]) : ((c & 1) ? p0[1] : p0[0]));
        float l1 =
            (c & 4) ? ((c & 2) ? ((c & 1) ? p1[7] : p1[6]) : ((c & 1) ? p1[5] : p1[4]))
                    : ((c & 2) ? ((c & 1) ? p1[3] : p1[2]) : ((c & 1) ? p1[1] : p1[0]));
        const long row = c + hi * 8;
        emit_edge(e0 + row,      l0 + b2v, u_gate, u_bern, out, n_edges);
        emit_edge(e0 + 16 + row, l1 + b2v, u_gate, u_bern, out, n_edges);
    }
}

extern "C" void kernel_launch(void* const* d_in, const int* in_sizes, int n_in,
                              void* d_out, int out_size, void* d_ws, size_t ws_size,
                              hipStream_t stream)
{
    const float* node_emb = (const float*)d_in[0];
    const int*   src      = (const int*)  d_in[1];
    const int*   dst      = (const int*)  d_in[2];
    const float* W1       = (const float*)d_in[3];
    const float* b1       = (const float*)d_in[4];
    const float* W2       = (const float*)d_in[5];
    const float* b2       = (const float*)d_in[6];
    const float* u_gate   = (const float*)d_in[7];
    const float* u_bern   = (const float*)d_in[8];
    float*       out      = (float*)d_out;
    _Float16*    w1p      = (_Float16*)d_ws;       // 64 KB packed W1

    (void)n_in; (void)out_size; (void)ws_size;

    pack_w1_kernel<<<SLOTS / 256, 256, 0, stream>>>(W1, w1p);

    int n_edges = in_sizes[1];                      // 800000
    int epb     = 256;                              // 8 waves * 32 edges
    int blocks  = (n_edges + epb - 1) / epb;        // 3125
    edge_gate_kernel<<<blocks, 256, 0, stream>>>(node_emb, src, dst, w1p,
                                                 b1, W2, b2, u_gate, u_bern,
                                                 out, n_edges);
}
